// BinaryDecoderWithRegularization_42734924595809
// MI455X (gfx1250) — compile-verified, matrix operand order checked by reading, not run on hardware
//
#include <hip/hip_runtime.h>
#include <hip/hip_bf16.h>

typedef _Float16 half_t;
typedef _Float16 v4h  __attribute__((ext_vector_type(4)));
typedef _Float16 v8h  __attribute__((ext_vector_type(8)));
typedef _Float16 v16h __attribute__((ext_vector_type(16)));
typedef float    v8f  __attribute__((ext_vector_type(8)));

#define IN_F   4096
#define OUT_F  768
#define NBITS  8
#define BATCH  1024
#define WB     (OUT_F * NBITS)     // 6144
#define TILES_N32 (OUT_F / 32)     // 24
#define TILES_M32 (BATCH / 32)     // 32

__device__ __forceinline__ float sigf(float x) { return 1.0f / (1.0f + __expf(-x)); }

// ---------------------------------------------------------------- init
__global__ void k_zero(float* acc) {
    acc[0] = 0.0f;  // recon sum
    acc[1] = 0.0f;  // reg sum
}

// ------------------------------------------- weight -> int_weights^T (f16) + reg partial
// one thread per (i, o); reads 8 contiguous f32 (coalesced 32B/thread),
// writes Wt[o*4096 + i] so the GEMM B-fragment is contiguous in K.
__global__ void __launch_bounds__(256)
k_prep_weights(const float* __restrict__ w, half_t* __restrict__ wt, float* __restrict__ acc) {
    int idx = blockIdx.x * 256 + threadIdx.x;          // < 4096*768
    int i = idx / OUT_F;
    int o = idx - i * OUT_F;

    const float4* wp = (const float4*)(w + (size_t)i * WB + (size_t)o * NBITS);
    float4 a = wp[0];
    float4 b = wp[1];

    float p0 = sigf(a.x), p1 = sigf(a.y), p2 = sigf(a.z), p3 = sigf(a.w);
    float p4 = sigf(b.x), p5 = sigf(b.y), p6 = sigf(b.z), p7 = sigf(b.w);

    float iw = p0 + 2.0f*p1 + 4.0f*p2 + 8.0f*p3
             + 16.0f*p4 + 32.0f*p5 + 64.0f*p6 - 128.0f*p7;

    float r = fminf(p0, 1.0f - p0) + fminf(p1, 1.0f - p1)
            + fminf(p2, 1.0f - p2) + fminf(p3, 1.0f - p3)
            + fminf(p4, 1.0f - p4) + fminf(p5, 1.0f - p5)
            + fminf(p6, 1.0f - p6) + fminf(p7, 1.0f - p7);

    wt[(size_t)o * IN_F + i] = (half_t)iw;

    __shared__ float red[256];
    red[threadIdx.x] = r;
    __syncthreads();
    for (int s = 128; s > 0; s >>= 1) {
        if (threadIdx.x < s) red[threadIdx.x] += red[threadIdx.x + s];
        __syncthreads();
    }
    if (threadIdx.x == 0) atomicAdd(&acc[1], red[0]);
}

// ---------------------------------------------------------------- latent f32 -> f16
__global__ void __launch_bounds__(256)
k_cvt_latent(const float* __restrict__ l, half_t* __restrict__ l16) {
    int idx = blockIdx.x * 256 + threadIdx.x;          // < 1024*4096/4
    float4 v = ((const float4*)l)[idx];
    v4h h = { (_Float16)v.x, (_Float16)v.y, (_Float16)v.z, (_Float16)v.w };
    ((v4h*)l16)[idx] = h;
}

// fragment loader: CDNA5 16-bit A/B layout, contiguous K per lane
//   lane<16 : K = 0..7 (VGPR0-3) and 16..23 (VGPR4-7)
//   lane>=16: K = 8..15 and 24..31
__device__ __forceinline__ v16h load_frag(const half_t* p) {
    v8h lo = *(const v8h*)(p);
    v8h hi = *(const v8h*)(p + 16);
    return __builtin_shufflevector(lo, hi, 0,1,2,3,4,5,6,7,8,9,10,11,12,13,14,15);
}

// ---------------------------------------------------------------- WMMA GEMM + fused recon loss
// One wave computes a 32x32 output block (2x2 WMMA tiles): per K-step,
// 2 A-fragments + 2 B-fragments feed 4 v_wmma_f32_16x16x32_f16.
// A = latent_f16 [1024 x 4096] row-major, B = Wt [768 x 4096] (int_weights^T).
__global__ void __launch_bounds__(256)
k_gemm_loss(const half_t* __restrict__ A16, const half_t* __restrict__ B16,
            const float* __restrict__ true_sum, float* __restrict__ acc) {
    int wave = threadIdx.x >> 5;
    int lane = threadIdx.x & 31;
    int tile = blockIdx.x * 8 + wave;                  // 0..767
    int mt = (tile / TILES_N32) << 5;                  // 32-row block
    int nt = (tile % TILES_N32) << 5;                  // 32-col block

    int row = lane & 15;
    int kb  = (lane >> 4) << 3;                        // 0 or 8

    const half_t* ap0 = A16 + (size_t)(mt + row)      * IN_F + kb;
    const half_t* ap1 = A16 + (size_t)(mt + 16 + row) * IN_F + kb;
    const half_t* bp0 = B16 + (size_t)(nt + row)      * IN_F + kb;
    const half_t* bp1 = B16 + (size_t)(nt + 16 + row) * IN_F + kb;

    v8f c00 = {0.f,0.f,0.f,0.f,0.f,0.f,0.f,0.f};
    v8f c01 = c00, c10 = c00, c11 = c00;

#pragma unroll 2
    for (int k = 0; k < IN_F; k += 32) {
        v16h a0 = load_frag(ap0 + k);
        v16h a1 = load_frag(ap1 + k);
        v16h b0 = load_frag(bp0 + k);
        v16h b1 = load_frag(bp1 + k);
        c00 = __builtin_amdgcn_wmma_f32_16x16x32_f16(false, a0, false, b0, (short)0, c00, false, false);
        c01 = __builtin_amdgcn_wmma_f32_16x16x32_f16(false, a0, false, b1, (short)0, c01, false, false);
        c10 = __builtin_amdgcn_wmma_f32_16x16x32_f16(false, a1, false, b0, (short)0, c10, false, false);
        c11 = __builtin_amdgcn_wmma_f32_16x16x32_f16(false, a1, false, b1, (short)0, c11, false, false);
    }

    // C/D layout: VGPR r -> M = r (lanes 0-15) or 8+r (lanes 16-31); N = lane&15
    int ncol  = lane & 15;
    int moff  = (lane >> 4) << 3;                      // 0 or 8
    float local = 0.0f;

    const v8f* cs[4] = { &c00, &c01, &c10, &c11 };
#pragma unroll
    for (int q = 0; q < 4; ++q) {
        int m0 = mt + ((q >> 1) << 4) + moff;
        int n  = nt + ((q & 1) << 4) + ncol;
        const v8f cq = *cs[q];
#pragma unroll
        for (int r = 0; r < 8; ++r) {
            int m = m0 + r;
            const float4* tp = (const float4*)(true_sum + (size_t)m * WB + (size_t)n * NBITS);
            float4 t0 = tp[0];
            float4 t1 = tp[1];
            float s = t0.x + 2.0f*t0.y + 4.0f*t0.z + 8.0f*t0.w
                    + 16.0f*t1.x + 32.0f*t1.y + 64.0f*t1.z - 128.0f*t1.w;
            float d = (cq[r] - s) * (1.0f / 255.0f);
            local += d * d;
        }
    }

    __shared__ float red[256];
    red[threadIdx.x] = local;
    __syncthreads();
    for (int s2 = 128; s2 > 0; s2 >>= 1) {
        if (threadIdx.x < s2) red[threadIdx.x] += red[threadIdx.x + s2];
        __syncthreads();
    }
    if (threadIdx.x == 0) atomicAdd(&acc[0], red[0]);
}

// ---------------------------------------------------------------- finalize
__global__ void k_final(const float* __restrict__ acc, float* __restrict__ out) {
    float recon = acc[0] * (1.0f / (float)(BATCH * OUT_F));            // /786432
    float reg   = acc[1] * (0.001f / (float)((size_t)IN_F * WB));      // /25165824
    out[0] = recon + reg;
    out[1] = recon;
    out[2] = reg;
}

// ---------------------------------------------------------------- launcher
extern "C" void kernel_launch(void* const* d_in, const int* in_sizes, int n_in,
                              void* d_out, int out_size, void* d_ws, size_t ws_size,
                              hipStream_t stream) {
    const float* latent   = (const float*)d_in[0];   // [1024, 4096]
    const float* true_sum = (const float*)d_in[1];   // [1024, 6144]
    const float* weight   = (const float*)d_in[2];   // [4096, 6144]
    float* out = (float*)d_out;                      // 3 scalars

    char* wsb = (char*)d_ws;
    float*  acc = (float*)wsb;                                         // 2 floats
    half_t* wt  = (half_t*)(wsb + 256);                                // 768*4096 f16 (6 MB)
    half_t* l16 = (half_t*)(wsb + 256 + (size_t)OUT_F * IN_F * 2);     // 1024*4096 f16 (8 MB)

    k_zero<<<1, 1, 0, stream>>>(acc);

    // 4096*768 threads, 256/block
    k_prep_weights<<<(IN_F * OUT_F) / 256, 256, 0, stream>>>(weight, wt, acc);

    // 1024*4096/4 threads
    k_cvt_latent<<<(BATCH * IN_F / 4) / 256, 256, 0, stream>>>(latent, l16);

    // 32*24 = 768 waves (32x32 block each), 8 waves/block -> 96 blocks
    k_gemm_loss<<<(TILES_M32 * TILES_N32) / 8, 256, 0, stream>>>(l16, wt, true_sum, acc);

    k_final<<<1, 1, 0, stream>>>(acc, out);
}
